// HashNeRF_77532749627709
// MI455X (gfx1250) — matrix operand order repaired and checked by reading, not compiled, not run on hardware
//
#include <hip/hip_runtime.h>
#include <math.h>

typedef __attribute__((ext_vector_type(16))) _Float16 v16h;
typedef __attribute__((ext_vector_type(8)))  _Float16 v8h;
typedef __attribute__((ext_vector_type(8)))  float    v8f;

#define TABLE_SIZE 524288
#define LEVELS     8
#define NRAYS      32768
#define NSAMP      64
#define TOTAL      (NRAYS*NSAMP)      /* 2097152 */
#define INP        71
#define INP_PAD    96
#define HID        128
#define STEPF      0.03125f

// ---------------------------------------------------------------------------
// Weight prep: f32 -> f16, transposed to [N][K] (K contiguous) so each WMMA
// B-fragment is one contiguous 32B load per lane. W0 padded K 71->96, W2
// padded N 4->16 with zeros.
// ---------------------------------------------------------------------------
__global__ __launch_bounds__(256) void prep_weights(
    const float* __restrict__ W0, const float* __restrict__ W1,
    const float* __restrict__ W2,
    _Float16* __restrict__ w0t, _Float16* __restrict__ w1t,
    _Float16* __restrict__ w2t)
{
  int i = blockIdx.x * 256 + threadIdx.x;
  if (i < HID * INP_PAD) {                 // w0t[h][k] = W0[k][h]
    int h = i / INP_PAD, k = i % INP_PAD;
    w0t[i] = (_Float16)(k < INP ? W0[k * HID + h] : 0.0f);
  }
  if (i < HID * HID) {                     // w1t[h2][h1] = W1[h1][h2]
    int h2 = i / HID, h1 = i % HID;
    w1t[i] = (_Float16)W1[h1 * HID + h2];
  }
  if (i < 16 * HID) {                      // w2t[n][h] = W2[h][n] (n<4)
    int n = i / HID, h = i % HID;
    w2t[i] = (_Float16)(n < 4 ? W2[h * 4 + n] : 0.0f);
  }
}

// ---------------------------------------------------------------------------
// Main fused kernel: 128 threads = 4 waves; 128 samples per workgroup.
// ---------------------------------------------------------------------------
__global__ __launch_bounds__(128) void nerf_kernel(
    const float* __restrict__ ray_o, const float* __restrict__ ray_d,
    const float* __restrict__ tables,
    const float* __restrict__ b0, const float* __restrict__ b1,
    const float* __restrict__ b2,
    const _Float16* __restrict__ w0t, const _Float16* __restrict__ w1t,
    const _Float16* __restrict__ w2t,
    float* __restrict__ out)
{
  __shared__ _Float16 sA[128 * INP_PAD];    // 8 M-tiles of [16][96]  (24 KB)
  __shared__ _Float16 sAct[4 * 16 * HID];   // per-wave [16][128]     (16 KB)

  const int t      = threadIdx.x;
  const int wgBase = blockIdx.x * 128;

  // ---------------- Phase 1: per-sample feature vector -> LDS (f16) --------
  {
    int s   = wgBase + t;
    int ray = s >> 6;
    int si  = s & 63;
    float ox = ray_o[ray * 3 + 0], oy = ray_o[ray * 3 + 1], oz = ray_o[ray * 3 + 2];
    float dx = ray_d[ray * 3 + 0], dy = ray_d[ray * 3 + 1], dz = ray_d[ray * 3 + 2];
    float tt = (float)si * STEPF;
    float px = ox + tt * dx, py = oy + tt * dy, pz = oz + tt * dz;
    // clamp to volume
    float cx = fminf(fmaxf(px, -1.0f), 1.0f);
    float cy = fminf(fmaxf(py, -1.0f), 1.0f);
    float cz = fminf(fmaxf(pz, -0.5f), 1.5f);

    _Float16* row = &sA[(t >> 4) * (16 * INP_PAD) + (t & 15) * INP_PAD];

    #pragma unroll
    for (int l = 0; l < LEVELS; l++) {
      float sl  = 0.25f / (float)(l + 1);
      float inv = (float)(l + 1) * 4.0f;
      float bx = floorf(cx * inv) * sl;
      float by = floorf(cy * inv) * sl;
      float bz = floorf(cz * inv) * sl;
      float wx1 = (cx - bx) * inv, wx0 = 1.0f - wx1;
      float wy1 = (cy - by) * inv, wy0 = 1.0f - wy1;
      float wz1 = (cz - bz) * inv, wz0 = 1.0f - wz1;
      float f0 = 0.f, f1 = 0.f, f2 = 0.f, f3 = 0.f;
      #pragma unroll
      for (int c = 0; c < 8; c++) {
        float vx = bx + 0.25f * (float)(c & 1);
        float vy = by + 0.25f * (float)((c >> 1) & 1);
        float vz = bz + 0.25f * (float)((c >> 2) & 1);
        float h = fmodf(vx * 73856093.0f + vy * 19349663.0f + vz * 83492791.0f,
                        524288.0f);
        h = (h < 0.0f) ? h + 524288.0f : h;
        int idx = (int)h;
        const float* tp = tables + ((size_t)l * TABLE_SIZE + (size_t)idx) * 4;
        float w = ((c & 1) ? wx1 : wx0) * (((c >> 1) & 1) ? wy1 : wy0) *
                  (((c >> 2) & 1) ? wz1 : wz0);
        f0 += w * tp[0]; f1 += w * tp[1]; f2 += w * tp[2]; f3 += w * tp[3];
      }
      row[l * 4 + 0] = (_Float16)f0;
      row[l * 4 + 1] = (_Float16)f1;
      row[l * 4 + 2] = (_Float16)f2;
      row[l * 4 + 3] = (_Float16)f3;
    }

    // direction positional encoding (39 values at [32..70])
    const float PI = 3.14159265358979f;
    float v3[3] = {dx, dy, dz};
    #pragma unroll
    for (int p = 0; p < 6; p++) {
      float mul = (p == 0) ? PI : (2.0f * PI * (float)p);
      #pragma unroll
      for (int c2 = 0; c2 < 3; c2++) {
        float ang = mul * v3[c2];
        row[32 + p * 6 + c2]     = (_Float16)__sinf(ang);
        row[32 + p * 6 + 3 + c2] = (_Float16)__cosf(ang);
      }
    }
    row[68] = (_Float16)dx; row[69] = (_Float16)dy; row[70] = (_Float16)dz;
    #pragma unroll
    for (int k = INP; k < INP_PAD; k++) row[k] = (_Float16)0.0f;
  }
  __syncthreads();

  // ---------------- Phase 2: WMMA MLP, 2 M-tiles per wave ------------------
  const int wave = t >> 5;
  const int lane = t & 31;
  const int hi   = lane >> 4;     // half-wave select
  const int ln   = lane & 15;     // N column / A row within tile
  _Float16* act  = &sAct[wave * 16 * HID];

  const v8f vzero = {0.f, 0.f, 0.f, 0.f, 0.f, 0.f, 0.f, 0.f};

  for (int r = 0; r < 2; r++) {
    const int lt = wave * 2 + r;
    const _Float16* Atile = &sA[lt * 16 * INP_PAD];
    const int sBase = wgBase + lt * 16;

    v8f acc[8];
    #pragma unroll
    for (int j = 0; j < 8; j++) acc[j] = vzero;

    // ---- layer 0: [16x96] x [96x128] ----
    #pragma unroll
    for (int k = 0; k < 3; k++) {
      v16h a;
      const _Float16* ap = Atile + ln * INP_PAD + k * 32 + hi * 16;
      ((v8h*)&a)[0] = *(const v8h*)(ap);
      ((v8h*)&a)[1] = *(const v8h*)(ap + 8);
      #pragma unroll
      for (int j = 0; j < 8; j++) {
        v16h b = *(const v16h*)(w0t + (j * 16 + ln) * INP_PAD + k * 32 + hi * 16);
        acc[j] = __builtin_amdgcn_wmma_f32_16x16x32_f16(
            false, a, false, b, (short)0, acc[j], false, false);
      }
    }
    // bias + relu -> f16 activation tile in LDS (A layout for next layer)
    #pragma unroll
    for (int j = 0; j < 8; j++) {
      float bias = b0[j * 16 + ln];
      #pragma unroll
      for (int v = 0; v < 8; v++) {
        float x = fmaxf(acc[j][v] + bias, 0.0f);
        act[(v + hi * 8) * HID + j * 16 + ln] = (_Float16)x;
      }
      acc[j] = vzero;
    }
    asm volatile("s_wait_dscnt 0" ::: "memory");  // cross-lane LDS handoff

    // ---- layer 1: [16x128] x [128x128] ----
    #pragma unroll
    for (int k = 0; k < 4; k++) {
      v16h a;
      const _Float16* ap = act + ln * HID + k * 32 + hi * 16;
      ((v8h*)&a)[0] = *(const v8h*)(ap);
      ((v8h*)&a)[1] = *(const v8h*)(ap + 8);
      #pragma unroll
      for (int j = 0; j < 8; j++) {
        v16h b = *(const v16h*)(w1t + (j * 16 + ln) * HID + k * 32 + hi * 16);
        acc[j] = __builtin_amdgcn_wmma_f32_16x16x32_f16(
            false, a, false, b, (short)0, acc[j], false, false);
      }
    }
    #pragma unroll
    for (int j = 0; j < 8; j++) {
      float bias = b1[j * 16 + ln];
      #pragma unroll
      for (int v = 0; v < 8; v++) {
        float x = fmaxf(acc[j][v] + bias, 0.0f);
        act[(v + hi * 8) * HID + j * 16 + ln] = (_Float16)x;
      }
    }
    asm volatile("s_wait_dscnt 0" ::: "memory");

    // ---- layer 2: [16x128] x [128x16] (cols 4..15 are zero pad) ----
    v8f acc2 = vzero;
    #pragma unroll
    for (int k = 0; k < 4; k++) {
      v16h a;
      const _Float16* ap = act + ln * HID + k * 32 + hi * 16;
      ((v8h*)&a)[0] = *(const v8h*)(ap);
      ((v8h*)&a)[1] = *(const v8h*)(ap + 8);
      v16h b = *(const v16h*)(w2t + ln * HID + k * 32 + hi * 16);
      acc2 = __builtin_amdgcn_wmma_f32_16x16x32_f16(
          false, a, false, b, (short)0, acc2, false, false);
    }

    // sigmoid + scatter: n==0 -> density, n in 1..3 -> rgb
    if (ln < 4) {
      float bias = b2[ln];
      #pragma unroll
      for (int v = 0; v < 8; v++) {
        int m = v + hi * 8;
        int sample = sBase + m;
        float x = acc2[v] + bias;
        float y = 1.0f / (1.0f + __expf(-x));
        if (ln == 0) out[sample] = y;
        else         out[TOTAL + sample * 3 + (ln - 1)] = y;
      }
    }
  }
}

// ---------------------------------------------------------------------------
extern "C" void kernel_launch(void* const* d_in, const int* in_sizes, int n_in,
                              void* d_out, int out_size, void* d_ws, size_t ws_size,
                              hipStream_t stream) {
  const float* ray_o  = (const float*)d_in[0];
  const float* ray_d  = (const float*)d_in[1];
  const float* tables = (const float*)d_in[2];
  const float* W0     = (const float*)d_in[3];
  const float* b0     = (const float*)d_in[4];
  const float* W1     = (const float*)d_in[5];
  const float* b1     = (const float*)d_in[6];
  const float* W2     = (const float*)d_in[7];
  const float* b2     = (const float*)d_in[8];
  float* out = (float*)d_out;

  // workspace layout (all offsets 32B-aligned for v16h loads)
  _Float16* w0t = (_Float16*)d_ws;                             // 128*96  f16
  _Float16* w1t = (_Float16*)((char*)d_ws + 24576);            // 128*128 f16
  _Float16* w2t = (_Float16*)((char*)d_ws + 24576 + 32768);    // 16*128  f16

  prep_weights<<<64, 256, 0, stream>>>(W0, W1, W2, w0t, w1t, w2t);
  nerf_kernel<<<TOTAL / 128, 128, 0, stream>>>(ray_o, ray_d, tables,
                                               b0, b1, b2, w0t, w1t, w2t, out);
}
